// DisneyKPAE_41549513622104
// MI455X (gfx1250) — matrix operand order repaired and checked by reading, not compiled, not run on hardware
//
#include <hip/hip_runtime.h>
#include <hip/hip_fp16.h>

// ============================================================================
// Disney KPCN denoiser for gfx1250 (MI455X).
// Conv chains run as implicit-GEMM on v_wmma_f32_16x16x32_f16 (f16 in, f32 acc).
// Activations: f16, channel-last [pixel][Cin_pad]   (Cin_pad in {32,128})
// Weights:     f16, [tap(25)][Cout_pad][Cin_pad]    (Cout_pad multiple of 64)
// Workspace:   [0,16M) actA | [16M,32M) actB | [32M,~35M) wbuf |
//              [35.5M] biasPad | [36M,~97M) logits
// ============================================================================

typedef __attribute__((ext_vector_type(16))) _Float16 v16h;
typedef __attribute__((ext_vector_type(8)))  _Float16 v8h;
typedef __attribute__((ext_vector_type(8)))  float    v8f;
typedef __attribute__((ext_vector_type(4)))  float    v4f;
typedef __attribute__((ext_vector_type(4)))  unsigned v4u;

#define KS   21
#define K2   441
#define LDSA_DW 20   // dwords per A row in LDS (16 data + 4 pad, keeps 16B align)
#define LDSB_DW 20   // dwords per B row in LDS
#define LDS_ABUF (64 * LDSA_DW)
#define LDS_BBUF (64 * LDSB_DW)

// ---------------------------------------------------------------------------
// Repack f32 NCHW activations -> f16 [pixel][Cpad] (zero-padded channels)
// ---------------------------------------------------------------------------
__global__ __launch_bounds__(256) void pack_acts(
    const float* __restrict__ x, _Float16* __restrict__ act,
    int B, int C, int Cpad, int HW)
{
  int idx = blockIdx.x * 256 + threadIdx.x;
  int total = B * HW * Cpad;
  if (idx >= total) return;
  int c  = idx % Cpad;
  int p  = idx / Cpad;
  int b  = p / HW;
  int pl = p - b * HW;
  float v = (c < C) ? x[(size_t)(b * C + c) * HW + pl] : 0.0f;
  act[idx] = (_Float16)v;
}

// ---------------------------------------------------------------------------
// Repack f32 (Cout,Cin,5,5) weights -> f16 [tap][Cout_pad][Cin_pad]
// ---------------------------------------------------------------------------
__global__ __launch_bounds__(256) void pack_weights(
    const float* __restrict__ w, _Float16* __restrict__ w16,
    int Cout, int Cin, int CoutPad, int CinPad)
{
  int idx = blockIdx.x * 256 + threadIdx.x;
  int total = 25 * CoutPad * CinPad;
  if (idx >= total) return;
  int ci  = idx % CinPad;
  int r   = idx / CinPad;
  int co  = r % CoutPad;
  int tap = r / CoutPad;
  float v = (co < Cout && ci < Cin) ? w[(size_t)(co * Cin + ci) * 25 + tap] : 0.0f;
  w16[idx] = (_Float16)v;
}

// ---------------------------------------------------------------------------
// Zero-pad bias to [CoutPad] so the conv epilogue can load it branch-free.
// ---------------------------------------------------------------------------
__global__ __launch_bounds__(256) void pack_bias(
    const float* __restrict__ bias, float* __restrict__ biasPad,
    int CoutReal, int CoutPad)
{
  int idx = blockIdx.x * 256 + threadIdx.x;
  if (idx >= CoutPad) return;
  biasPad[idx] = (idx < CoutReal) ? bias[idx] : 0.0f;
}

// ---------------------------------------------------------------------------
// 5x5 VALID conv as implicit GEMM on WMMA.
//   D[co][pix] = sum_{tap,ci} W[tap][co][ci] * X[pix_in(tap)][ci]
// Block: 256 threads = 8 waves. Block tile: M=64 (4 wmma tiles) x N=64 pixels
// (4 wmma tiles). Wave w -> mSub = w&3 (M tile), nPair = w>>2 (N tiles
// 2*nPair, 2*nPair+1): one A fragment feeds two WMMAs per K step.
// K loop: flattened 25 taps x (CIN_PAD/32) tiles, software-pipelined with
// double-buffered LDS (one barrier per step). Staging: each thread owns one
// contiguous 16B segment per matrix -> global_load_b128 + ds_store_b128.
// All addressing is 32-bit adds on precomputed per-thread dword offsets;
// CIN_PAD is a template constant so strides are shifts, not v_mul_u64.
// MODE 0: bias + ReLU -> f16 actOut[pix][CoutPad]
// MODE 1: bias        -> f32 logitOut[pix][CoutPad]
// ---------------------------------------------------------------------------
template <int CIN_PAD, int MODE>
__global__ __launch_bounds__(256) void conv5x5_wmma(
    const _Float16* __restrict__ actIn,
    const _Float16* __restrict__ w16,
    const float*    __restrict__ biasPad,   // [CoutPad], zero-padded
    _Float16*       __restrict__ actOut,
    float*          __restrict__ logitOut,
    int B, int Hin, int Win,
    int CoutPad)
{
  constexpr int CIN_DW   = CIN_PAD / 2;   // dwords per activation row (16 or 64)
  constexpr int KT_TILES = CIN_PAD / 32;  // K tiles per tap (1 or 4)

  const int Hout  = Hin - 4, Wout = Win - 4;
  const int HWout = Hout * Wout;
  const int Npix  = B * HWout;
  const int HWin  = Hin * Win;

  __shared__ unsigned AsU[2 * LDS_ABUF];  // double-buffered 64x32-half A tiles
  __shared__ unsigned BsU[2 * LDS_BBUF];  // double-buffered 64x32-half B tiles

  const int t     = threadIdx.x;
  const int lane  = t & 31;
  const int wave  = t >> 5;
  const int mSub  = wave & 3;     // M tile
  const int nPair = wave >> 2;    // N tile pair (0 or 1)
  const int g     = lane >> 4;    // lane group (K split)
  const int ln    = lane & 15;    // M (A) / N (B) index within tile

  const int coBlk  = blockIdx.y * 64;
  const int pixBlk = blockIdx.x * 64;

  const unsigned* wU = (const unsigned*)w16;
  const unsigned* aU = (const unsigned*)actIn;

  // --- per-thread staging assignment: one 16B segment of one row, per matrix
  const int row = t >> 2;          // 0..63
  const int seg = (t & 3) * 4;     // dword offset of 16B segment within row

  const int wBase = (coBlk + row) * CIN_DW + seg;        // A global (dwords)
  int n = pixBlk + row;
  n = (n < Npix) ? n : (Npix - 1);                       // clamp; stores guarded
  int bb = n / HWout; int rr = n - bb * HWout;
  int yy = rr / Wout; int xx = rr - yy * Wout;
  const int pBase = (bb * HWin + yy * Win + xx) * CIN_DW + seg;  // B global

  const int ldsA = row * LDSA_DW + seg;                  // LDS store offsets
  const int ldsB = row * LDSB_DW + seg;

  // --- fragment read offsets in LDS (dwords) ---
  const int aFragOff = (mSub * 16 + ln) * LDSA_DW;
  const int bFragOff = (nPair * 32 + ln) * LDSB_DW;

  v8f acc0 = {}, acc1 = {};

  const int tapStrideW = CoutPad * CIN_DW;  // A advance per tap (dwords)
  int dA = 0, dB = 0, ktc = 0, kx = 0, cur = 0;

  auto stageStep = [&](unsigned* Ad, unsigned* Bd) {
    *(v4u*)(Ad + ldsA) = *(const v4u*)(wU + wBase + dA);
    *(v4u*)(Bd + ldsB) = *(const v4u*)(aU + pBase + dB);
    if (++ktc == KT_TILES) {                 // advance to next tap
      ktc = 0;
      dA += tapStrideW - (KT_TILES - 1) * 16;
      const int adv = (kx == 4) ? (Win - 4) * CIN_DW : CIN_DW;
      kx = (kx == 4) ? 0 : (kx + 1);
      dB += adv - (KT_TILES - 1) * 16;
    } else {                                 // next K tile within this tap
      dA += 16;
      dB += 16;
    }
  };

  const int NSTEPS = 25 * KT_TILES;
  stageStep(AsU, BsU);                       // prologue: fill buffer 0

  for (int step = 0; step < NSTEPS; ++step) {
    __syncthreads();                         // buffer `cur` ready; prior reads done
    if (step + 1 < NSTEPS) {                 // stage next tile into other buffer
      stageStep(AsU + (cur ^ 1) * LDS_ABUF, BsU + (cur ^ 1) * LDS_BBUF);
      // gfx1250 global_prefetch of the step-after-next activation segment
      __builtin_prefetch((const void*)(aU + pBase + dB), 0, 1);
    }

    const unsigned* As = AsU + cur * LDS_ABUF;
    const unsigned* Bs = BsU + cur * LDS_BBUF;
    const unsigned* Arow  = As + aFragOff;
    const unsigned* Brow0 = Bs + bFragOff;
    const unsigned* Brow1 = Brow0 + 16 * LDSB_DW;

    // A fragment (16x32 f16): lane row M=ln; vgpr p: K=16g+2p{,+1}
    //   -> 8 consecutive dwords starting at 8g
    union { unsigned u[8]; v16h v; } af, bf0, bf1;
    #pragma unroll
    for (int p = 0; p < 8; ++p) {
      const int dw = 8 * g + p;
      af.u[p] = Arow[dw];
    }
    // B fragment (32x16 f16): lane col N=ln; lanes 0-15 K=0-15, 16-31
    // K=16-31 -> 8 consecutive dwords starting at 8g
    #pragma unroll
    for (int p = 0; p < 8; ++p) {
      bf0.u[p] = Brow0[8 * g + p];
      bf1.u[p] = Brow1[8 * g + p];
    }

    acc0 = __builtin_amdgcn_wmma_f32_16x16x32_f16(
        false, af.v, false, bf0.v, (short)0, acc0, false, false);
    acc1 = __builtin_amdgcn_wmma_f32_16x16x32_f16(
        false, af.v, false, bf1.v, (short)0, acc1, false, false);

    cur ^= 1;
  }

  // --- epilogue: D layout -> lane ln = N (pixel), vgpr r -> M = 8g + r ---
  const int coBase = coBlk + mSub * 16 + 8 * g;      // multiple of 8
  const v4f bv0 = *(const v4f*)(biasPad + coBase);   // branch-free padded bias
  const v4f bv1 = *(const v4f*)(biasPad + coBase + 4);

  #pragma unroll
  for (int half = 0; half < 2; ++half) {
    v8f& acc = half ? acc1 : acc0;
    const int pix = pixBlk + (nPair * 2 + half) * 16 + ln;
    if (pix < Npix) {
      if (MODE == 0) {
        v8h hv;
        #pragma unroll
        for (int r = 0; r < 8; ++r) {
          float v = acc[r] + ((r < 4) ? bv0[r & 3] : bv1[r & 3]);
          hv[r] = (_Float16)(v > 0.0f ? v : 0.0f);   // ReLU
        }
        *(v8h*)(actOut + (size_t)pix * CoutPad + coBase) = hv;  // 16B store
      } else {
        float* o = logitOut + (size_t)pix * CoutPad + coBase;
        #pragma unroll
        for (int r = 0; r < 8; ++r)
          o[r] = acc[r] + ((r < 4) ? bv0[r & 3] : bv1[r & 3]);
      }
    }
  }
}

// ---------------------------------------------------------------------------
// In-place softmax over the first Creal channels of logits[pix][Cpad].
// One wave32 per pixel; shfl_xor reductions.
// ---------------------------------------------------------------------------
__global__ __launch_bounds__(256) void softmax_channels(
    float* __restrict__ logits, int Npix, int Cpad, int Creal)
{
  const int lane = threadIdx.x & 31;
  const int pix  = blockIdx.x * 8 + (threadIdx.x >> 5);
  if (pix >= Npix) return;
  float* row = logits + (size_t)pix * Cpad;

  float mx = -3.0e38f;
  for (int c = lane; c < Creal; c += 32) mx = fmaxf(mx, row[c]);
  #pragma unroll
  for (int off = 16; off > 0; off >>= 1) mx = fmaxf(mx, __shfl_xor(mx, off, 32));

  float sum = 0.0f;
  for (int c = lane; c < Creal; c += 32) {
    float e = __expf(row[c] - mx);
    row[c] = e;
    sum += e;
  }
  #pragma unroll
  for (int off = 16; off > 0; off >>= 1) sum += __shfl_xor(sum, off, 32);

  float inv = 1.0f / sum;
  for (int c = lane; c < Creal; c += 32) row[c] *= inv;
}

// ---------------------------------------------------------------------------
// Apply predicted 21x21 kernels: out[b,c,y,x] = sum_t probs[pix][t] *
//   buf[b,c, crop+y+ty, crop+x+tx]. probs layout [pix][448], pix=b*Ho*Ho+y*Ho+x.
// ---------------------------------------------------------------------------
__global__ __launch_bounds__(256) void apply_kernels(
    const float* __restrict__ probs, const float* __restrict__ buf,
    float* __restrict__ out, int B, int Hb, int Ho, int crop)
{
  int idx = blockIdx.x * 256 + threadIdx.x;
  int total = B * 3 * Ho * Ho;
  if (idx >= total) return;
  int x = idx % Ho;
  int y = (idx / Ho) % Ho;
  int c = (idx / (Ho * Ho)) % 3;
  int b = idx / (3 * Ho * Ho);
  int pix = (b * Ho + y) * Ho + x;
  const float* pr = probs + (size_t)pix * (K2 + 7);   // Cpad = 448
  const float* bp = buf + ((size_t)(b * 3 + c) * Hb + (crop + y)) * Hb + (crop + x);
  float s = 0.0f;
  #pragma unroll 3
  for (int ty = 0; ty < KS; ++ty) {
    const float* brow = bp + ty * Hb;
    #pragma unroll
    for (int tx = 0; tx < KS; ++tx)
      s += pr[ty * KS + tx] * brow[tx];
  }
  out[idx] = s;
}

// ---------------------------------------------------------------------------
// radiance = albedo_c * r_diffuse + (exp(r_specular) - 1)
// ---------------------------------------------------------------------------
__global__ __launch_bounds__(256) void combine_out(
    const float* __restrict__ rd, const float* __restrict__ rs,
    const float* __restrict__ albedo, float* __restrict__ rad,
    int B, int Hb, int Ho, int off)
{
  int idx = blockIdx.x * 256 + threadIdx.x;
  int total = B * 3 * Ho * Ho;
  if (idx >= total) return;
  int x = idx % Ho;
  int y = (idx / Ho) % Ho;
  int c = (idx / (Ho * Ho)) % 3;
  int b = idx / (3 * Ho * Ho);
  float a = albedo[((size_t)(b * 3 + c) * Hb + (off + y)) * Hb + (off + x)];
  rad[idx] = a * rd[idx] + (__expf(rs[idx]) - 1.0f);
}

// ===========================================================================
extern "C" void kernel_launch(void* const* d_in, const int* in_sizes, int n_in,
                              void* d_out, int out_size, void* d_ws, size_t ws_size,
                              hipStream_t stream)
{
  (void)in_sizes; (void)n_in; (void)out_size; (void)ws_size;
  const int B = 4, HIMG = 128, HOUT = 92;

  const float* diffuse_in      = (const float*)d_in[0];
  const float* specular_in     = (const float*)d_in[1];
  const float* diffuse_buffer  = (const float*)d_in[2];
  const float* specular_buffer = (const float*)d_in[3];
  const float* albedo          = (const float*)d_in[4];
  const float* dw0 = (const float*)d_in[5];
  const float* db0 = (const float*)d_in[6];
  const float* dwh = (const float*)d_in[7];
  const float* dbh = (const float*)d_in[8];
  const float* dwo = (const float*)d_in[9];
  const float* dbo = (const float*)d_in[10];
  const float* sw0 = (const float*)d_in[11];
  const float* sb0 = (const float*)d_in[12];
  const float* swh = (const float*)d_in[13];
  const float* sbh = (const float*)d_in[14];
  const float* swo = (const float*)d_in[15];
  const float* sbo = (const float*)d_in[16];

  // workspace carve-up (~97 MB needed)
  char* ws = (char*)d_ws;
  _Float16* actA    = (_Float16*)(ws);
  _Float16* actB    = (_Float16*)(ws + ((size_t)16 << 20));
  _Float16* wbuf    = (_Float16*)(ws + ((size_t)32 << 20));   // <= 2.87 MB
  float*    biasPad = (float*)   (ws + ((size_t)35 << 20) + (512 << 10));
  float*    logits  = (float*)   (ws + ((size_t)36 << 20));

  float* out = (float*)d_out;
  const int planeN = B * 3 * HOUT * HOUT;
  float* rad = out;
  float* rd  = out + planeN;
  float* rs  = rd + planeN;

  auto runChain = [&](const float* xin, const float* w0, const float* b0,
                      const float* wh, const float* bh,
                      const float* wo, const float* bo,
                      const float* buf, float* rOut) {
    int Hin = HIMG;
    // input: 27 -> pad 32 channels, f16 channel-last
    { int n = B * Hin * Hin * 32;
      pack_acts<<<(n + 255) / 256, 256, 0, stream>>>(xin, actA, B, 27, 32, Hin * Hin); }
    // layer 0: Cin_pad 32 -> Cout_pad 128 (real 100)
    { int n = 25 * 128 * 32;
      pack_weights<<<(n + 255) / 256, 256, 0, stream>>>(w0, wbuf, 100, 27, 128, 32); }
    pack_bias<<<1, 256, 0, stream>>>(b0, biasPad, 100, 128);
    { int Ho = Hin - 4, Npix = B * Ho * Ho;
      dim3 grid((Npix + 63) / 64, 128 / 64, 1);
      conv5x5_wmma<32, 0><<<grid, 256, 0, stream>>>(actA, wbuf, biasPad, actB,
                                                    nullptr, B, Hin, Hin, 128); }
    Hin -= 4;

    _Float16* src = actB; _Float16* dst = actA;
    for (int l = 0; l < 7; ++l) {
      int n = 25 * 128 * 128;
      pack_weights<<<(n + 255) / 256, 256, 0, stream>>>(
          wh + (size_t)l * 100 * 100 * 25, wbuf, 100, 100, 128, 128);
      pack_bias<<<1, 256, 0, stream>>>(bh + l * 100, biasPad, 100, 128);
      int Ho = Hin - 4, Npix = B * Ho * Ho;
      dim3 grid((Npix + 63) / 64, 128 / 64, 1);
      conv5x5_wmma<128, 0><<<grid, 256, 0, stream>>>(src, wbuf, biasPad, dst,
                                                     nullptr, B, Hin, Hin, 128);
      Hin -= 4;
      _Float16* tmp = src; src = dst; dst = tmp;
    }

    // output layer: Cin_pad 128 -> Cout_pad 448 (real 441), f32 logits
    { int n = 25 * 448 * 128;
      pack_weights<<<(n + 255) / 256, 256, 0, stream>>>(wo, wbuf, 441, 100, 448, 128); }
    pack_bias<<<2, 256, 0, stream>>>(bo, biasPad, 441, 448);
    { int Ho = Hin - 4, Npix = B * Ho * Ho;
      dim3 grid((Npix + 63) / 64, 448 / 64, 1);
      conv5x5_wmma<128, 1><<<grid, 256, 0, stream>>>(src, wbuf, biasPad, nullptr,
                                                     logits, B, Hin, Hin, 448); }
    Hin -= 4;                       // = 92

    int Npix = B * Hin * Hin;
    softmax_channels<<<(Npix + 7) / 8, 256, 0, stream>>>(logits, Npix, 448, 441);
    int nOut = B * 3 * Hin * Hin;
    apply_kernels<<<(nOut + 255) / 256, 256, 0, stream>>>(logits, buf, rOut,
                                                          B, HIMG, Hin, 8);
  };

  runChain(diffuse_in,  dw0, db0, dwh, dbh, dwo, dbo, diffuse_buffer,  rd);
  runChain(specular_in, sw0, sb0, swh, sbh, swo, sbo, specular_buffer, rs);
  combine_out<<<(planeN + 255) / 256, 256, 0, stream>>>(rd, rs, albedo, rad,
                                                        B, HIMG, HOUT,
                                                        (HIMG - HOUT) / 2);
}
